// MS_MSA_27986006901268
// MI455X (gfx1250) — compile-verified, matrix operand order checked by reading, not run on hardware
//
#include <hip/hip_runtime.h>
#include <hip/hip_bf16.h>
#include <math.h>

#define DIMC 112
#define KP   128
#define HEADS 4
#define DHEAD 28
#define HH 256
#define WW 256
#define WP2 478
#define NPIX (HH*WW)           // 65536
#define BATCH 4
#define NROWS (BATCH*NPIX)     // 262144
#define GRAM_SLICES 128

typedef __bf16 bf16;
typedef __attribute__((ext_vector_type(16))) __bf16 v16bf;
typedef __attribute__((ext_vector_type(8)))  float  v8f;
typedef __attribute__((ext_vector_type(4)))  unsigned int v4u;
typedef __attribute__((ext_vector_type(8)))  int v8i;
typedef __attribute__((ext_vector_type(4)))  int v4i;

#ifndef __has_builtin
#define __has_builtin(x) 0
#endif
#if __has_builtin(__builtin_amdgcn_tensor_load_to_lds) && \
    __has_builtin(__builtin_amdgcn_s_wait_tensorcnt)
#define USE_TDM 1
#else
#define USE_TDM 0
#endif

static __device__ __forceinline__ unsigned short f2bfu(float f){
  union { float f; unsigned u; } c; c.f = f;
  unsigned r = c.u + 0x7FFFu + ((c.u >> 16) & 1u);
  return (unsigned short)(r >> 16);
}
static __device__ __forceinline__ bf16 f2bf(float f){
  union { unsigned short s; bf16 b; } o; o.s = f2bfu(f); return o.b;
}
static __device__ __forceinline__ float bf2f(bf16 b){
  union { bf16 b; unsigned short s; } i; i.b = b;
  union { unsigned u; float f; } o; o.u = ((unsigned)i.s) << 16; return o.f;
}
// 16-bit A/B fragment element j -> K offset in a 32-wide chunk (CDNA5 ISA layout):
// j<8:  K = hi*8 + j ; j>=8: K = 16 + hi*8 + (j-8)
static __device__ __forceinline__ int koffset(int j, int hi){
  int p = j >> 1, e = j & 1;
  return ((p < 4) ? 0 : 16) + hi * 8 + ((p & 3) << 1) + e;
}
// inverse: global k -> (kc, hi, j)
static __device__ __forceinline__ void kinv(int k, int& kc, int& hi, int& j){
  kc = k >> 5;
  int koff = k & 31;
  hi = (koff >> 3) & 1;
  int t = koff & 23;              // drop the hi bit
  j = (t < 8) ? t : (t - 8);
}

// ---------------- mask branch (VALU; ~6 GFLOP total) ----------------

__global__ __launch_bounds__(256) void conv1x1_kernel(
    const float* __restrict__ in, const float* __restrict__ w,
    const float* __restrict__ bias, float* __restrict__ out, int npix){
  int total = npix * DIMC;
  for (int idx = blockIdx.x*blockDim.x + threadIdx.x; idx < total;
       idx += gridDim.x*blockDim.x){
    int p = idx / DIMC, o = idx - p*DIMC;
    const float* row = in + (long)p * DIMC;
    const float* wr  = w + o * DIMC;
    float s = bias[o];
    #pragma unroll 8
    for (int i = 0; i < DIMC; ++i) s = fmaf(row[i], wr[i], s);
    out[idx] = s;
  }
}

// dw5x5(m2) -> sigmoid -> ms = m1*(1+sig), fused with shift_back (col = 2*ch + j)
__global__ __launch_bounds__(256) void maskattn_kernel(
    const float* __restrict__ m1, const float* __restrict__ m2,
    const float* __restrict__ dww, const float* __restrict__ dwb,
    float* __restrict__ mattn){
  int total = NPIX * DIMC;
  for (int idx = blockIdx.x*blockDim.x + threadIdx.x; idx < total;
       idx += gridDim.x*blockDim.x){
    int ch = idx % DIMC; int pix = idx / DIMC;
    int r = pix >> 8, j = pix & 255;
    int cc = 2*ch + j;                       // < 478 always
    float acc = dwb[ch];
    #pragma unroll
    for (int dy = 0; dy < 5; ++dy){
      int rr = r + dy - 2;
      if (rr < 0 || rr >= HH) continue;
      #pragma unroll
      for (int dx = 0; dx < 5; ++dx){
        int c2 = cc + dx - 2;
        if (c2 < 0 || c2 >= WP2) continue;
        acc = fmaf(m2[((long)rr*WP2 + c2)*DIMC + ch], dww[ch*25 + dy*5 + dx], acc);
      }
    }
    float s = 1.f / (1.f + __expf(-acc));
    mattn[idx] = m1[((long)r*WP2 + cc)*DIMC + ch] * (1.f + s);
  }
}

// ---------------- fused QKV projection (WMMA bf16) ----------------
// LDS tiles stored PRE-PACKED in fragment order: slot((row,kc,hi,j)) so that each
// lane's 16-element fragment is one contiguous 32B chunk -> single v16bf LDS load.
__global__ __launch_bounds__(256) void qkv_wmma_kernel(
    const float* __restrict__ x,
    const float* __restrict__ Wq, const float* __restrict__ Wk,
    const float* __restrict__ Wv, const float* __restrict__ mattn,
    bf16* __restrict__ qb, bf16* __restrict__ kb,
    bf16* __restrict__ vmb, bf16* __restrict__ vib,
    float* __restrict__ normq, float* __restrict__ normk){
  __shared__ bf16 sXp[128 * KP];        // 32 KB, packed [row][kc][hi][j]
  __shared__ bf16 sWp[4 * DIMC * 32];   // 28 KB, packed [kc][o][hi][j]
  const int tid = threadIdx.x, wave = tid >> 5, lane = tid & 31;
  const int hi = lane >> 4, ln = lane & 15;
  const long rowbase = (long)blockIdx.x * 128;
  const int b = (int)(rowbase >> 16);

  // stage x: float4 global loads (k-quad stays contiguous in packed j) -> b64 LDS stores
  for (int idx = tid; idx < 128*28; idx += 256){
    int r = idx / 28, q4 = idx - r*28;
    int k0 = q4 * 4;
    const float4 v = *reinterpret_cast<const float4*>(x + (rowbase + r)*DIMC + k0);
    int kc, h2, j0; kinv(k0, kc, h2, j0);
    unsigned long long pk = (unsigned long long)f2bfu(v.x)
                          | ((unsigned long long)f2bfu(v.y) << 16)
                          | ((unsigned long long)f2bfu(v.z) << 32)
                          | ((unsigned long long)f2bfu(v.w) << 48);
    *reinterpret_cast<unsigned long long*>(&sXp[(((r*4 + kc)*2 + h2)*16) + j0]) = pk;
  }
  // zero-pad k in [112,128): kc=3, j in [8,16), both hi
  for (int idx = tid; idx < 128*4; idx += 256){
    int r = idx >> 2, h2 = (idx >> 1) & 1, qd = idx & 1;
    *reinterpret_cast<unsigned long long*>(
        &sXp[(((r*4 + 3)*2 + h2)*16) + 8 + qd*4]) = 0ull;
  }

  for (int mat = 0; mat < 3; ++mat){
    const float* Wmat = (mat == 0) ? Wq : ((mat == 1) ? Wk : Wv);
    __syncthreads();
    // stage weights: float4 along o (coalesced b128) -> 4 scattered b16 LDS stores
    for (int idx = tid; idx < DIMC*28; idx += 256){
      int k = idx / 28, o4 = (idx - k*28) * 4;
      const float4 v = *reinterpret_cast<const float4*>(Wmat + k*DIMC + o4);
      int kc, h2, j; kinv(k, kc, h2, j);
      sWp[(((kc*DIMC) + o4    )*2 + h2)*16 + j] = f2bf(v.x);
      sWp[(((kc*DIMC) + o4 + 1)*2 + h2)*16 + j] = f2bf(v.y);
      sWp[(((kc*DIMC) + o4 + 2)*2 + h2)*16 + j] = f2bf(v.z);
      sWp[(((kc*DIMC) + o4 + 3)*2 + h2)*16 + j] = f2bf(v.w);
    }
    for (int idx = tid; idx < DIMC*4; idx += 256){   // zero-pad k>=112
      int o = idx >> 2, h2 = (idx >> 1) & 1, qd = idx & 1;
      *reinterpret_cast<unsigned long long*>(
          &sWp[(((3*DIMC) + o)*2 + h2)*16 + 8 + qd*4]) = 0ull;
    }
    __syncthreads();

    v8f acc[7];
    #pragma unroll
    for (int i = 0; i < 7; ++i) acc[i] = (v8f)(0.f);
    const int arow = wave*16 + ln;

    #pragma unroll
    for (int kc = 0; kc < 4; ++kc){
      const v16bf Af = *reinterpret_cast<const v16bf*>(
          &sXp[((arow*4 + kc)*2 + hi)*16]);
      #pragma unroll
      for (int nt = 0; nt < 7; ++nt){
        const v16bf Bf = *reinterpret_cast<const v16bf*>(
            &sWp[(((kc*DIMC) + nt*16 + ln)*2 + hi)*16]);
        acc[nt] = __builtin_amdgcn_wmma_f32_16x16x32_bf16(
            false, Af, false, Bf, (short)0, acc[nt], false, false);
      }
    }

    // D layout: lane column ch = nt*16+ln ; rows m = i + hi*8
    #pragma unroll
    for (int nt = 0; nt < 7; ++nt){
      int ch = nt*16 + ln;
      float ss = 0.f;
      #pragma unroll
      for (int i = 0; i < 8; ++i){
        long row = rowbase + wave*16 + hi*8 + i;
        float val = acc[nt][i];
        long o = row * DIMC + ch;
        if (mat == 0){ qb[o] = f2bf(val); ss += val*val; }
        else if (mat == 1){ kb[o] = f2bf(val); ss += val*val; }
        else {
          vib[o] = f2bf(val);
          int pix = (int)(row & (NPIX - 1));
          vmb[o] = f2bf(val * mattn[(long)pix*DIMC + ch]);
        }
      }
      if (mat == 0)      atomicAdd(&normq[b*DIMC + ch], ss);
      else if (mat == 1) atomicAdd(&normk[b*DIMC + ch], ss);
    }
  }
}

// ---------------- Gram G[b] = K^T Q (WMMA reduction over n) ----------------
// Double-buffered staging via the Tensor Data Mover (tensor_load_to_lds) when the
// builtin exists; manual staging fallback otherwise. 7 waves: wave = d-tile.
#if USE_TDM
static __device__ __forceinline__ void tdm_load_rows(unsigned lds_off,
                                                     const bf16* gptr){
  unsigned long long ga = (unsigned long long)(size_t)gptr;
  // D# group0: count=1, lds_addr, global_addr[56:0], type=2
  v4u g0 = { 1u, lds_off, (unsigned)(ga & 0xffffffffu),
             (unsigned)((ga >> 32) & 0x01ffffffu) | 0x80000000u };
  // D# group1: data_size=2B, tensor_dim0=112, tensor_dim1=NROWS,
  //            tile 112 x 32, dim0_stride=112
  v8i g1 = { (int)(1u << 16),
             (int)(112u << 16),
             0,
             (int)((NROWS >> 16) | (112u << 16)),
             32,
             112,
             0, 0 };
  v4i gz = {0, 0, 0, 0};
#if __clang_major__ >= 23
  v8i gz8 = {0, 0, 0, 0, 0, 0, 0, 0};
  __builtin_amdgcn_tensor_load_to_lds(g0, g1, gz, gz, gz8, 0);
#else
  __builtin_amdgcn_tensor_load_to_lds(g0, g1, gz, gz, 0);
#endif
}
#endif

__global__ __launch_bounds__(224) void gram_wmma_kernel(
    const bf16* __restrict__ qb, const bf16* __restrict__ kb, float* __restrict__ G){
  __shared__ bf16 sQ[2][32][DIMC];   // 14 KB (double-buffered)
  __shared__ bf16 sK[2][32][DIMC];
  const int tid = threadIdx.x, wave = tid >> 5, lane = tid & 31;
  const int hi = lane >> 4, ln = lane & 15;
  const int b = blockIdx.x / GRAM_SLICES;
  const int slice = blockIdx.x % GRAM_SLICES;
  const long base = (long)b * NPIX + (long)slice * (NPIX / GRAM_SLICES);
  const int NSTEP = (NPIX / GRAM_SLICES) / 32;   // 16

  v8f acc[7];
  #pragma unroll
  for (int i = 0; i < 7; ++i) acc[i] = (v8f)(0.f);

#if USE_TDM
  if (wave == 0){
    tdm_load_rows((unsigned)(unsigned long long)(size_t)&sQ[0][0][0], qb + base*DIMC);
    tdm_load_rows((unsigned)(unsigned long long)(size_t)&sK[0][0][0], kb + base*DIMC);
  }
#endif
  for (int step = 0; step < NSTEP; ++step){
    const int p = step & 1;
#if USE_TDM
    if (wave == 0){
      if (step + 1 < NSTEP){
        long rn = base + (long)(step + 1)*32;
        tdm_load_rows((unsigned)(unsigned long long)(size_t)&sQ[1-p][0][0], qb + rn*DIMC);
        tdm_load_rows((unsigned)(unsigned long long)(size_t)&sK[1-p][0][0], kb + rn*DIMC);
        __builtin_amdgcn_s_wait_tensorcnt(2);   // stage 'step' complete
      } else {
        __builtin_amdgcn_s_wait_tensorcnt(0);
      }
    }
    __syncthreads();
#else
    const long r0 = base + (long)step*32;
    __syncthreads();
    for (int idx = tid; idx < 32*DIMC; idx += 224){
      int r = idx / DIMC, c = idx - r*DIMC;
      sQ[p][r][c] = qb[(r0 + r)*DIMC + c];
      sK[p][r][c] = kb[(r0 + r)*DIMC + c];
    }
    __syncthreads();
#endif
    v16bf Af;                        // A[m=d][kk=n] = k[n][d]
    #pragma unroll
    for (int j = 0; j < 16; ++j) Af[j] = sK[p][koffset(j, hi)][wave*16 + ln];
    #pragma unroll
    for (int et = 0; et < 7; ++et){
      v16bf Bf;                      // B[kk=n][e] = q[n][e]
      #pragma unroll
      for (int j = 0; j < 16; ++j) Bf[j] = sQ[p][koffset(j, hi)][et*16 + ln];
      acc[et] = __builtin_amdgcn_wmma_f32_16x16x32_bf16(
          false, Af, false, Bf, (short)0, acc[et], false, false);
    }
    __syncthreads();   // all reads of buffer p done before it is refilled
  }
  #pragma unroll
  for (int et = 0; et < 7; ++et)
    #pragma unroll
    for (int i = 0; i < 8; ++i)
      atomicAdd(&G[((long)b*DIMC + wave*16 + hi*8 + i)*DIMC + et*16 + ln], acc[et][i]);
}

// ---------------- softmax(attn) + fold into W_eff = blockdiag(attn^T) @ proj_w ----
// W_eff is written already fragment-packed so out_wmma can straight-copy to LDS.
__global__ __launch_bounds__(128) void attn_weff_kernel(
    const float* __restrict__ G, const float* __restrict__ normq,
    const float* __restrict__ normk, const float* __restrict__ rescale,
    const float* __restrict__ proj_w, bf16* __restrict__ Weff){
  __shared__ float sA[HEADS][DHEAD][DHEAD];
  const int b = blockIdx.x, t = threadIdx.x;
  if (t < DIMC){
    int h = t / DHEAD, d = t - h*DHEAD;
    float nk = fmaxf(sqrtf(normk[b*DIMC + t]), 1e-12f);
    float rs = rescale[h];
    float logit[DHEAD]; float mx = -1e30f;
    #pragma unroll
    for (int e = 0; e < DHEAD; ++e){
      float nq = fmaxf(sqrtf(normq[b*DIMC + h*DHEAD + e]), 1e-12f);
      float l = G[((long)b*DIMC + t)*DIMC + h*DHEAD + e] / (nk*nq) * rs;
      logit[e] = l; mx = fmaxf(mx, l);
    }
    float sum = 0.f;
    #pragma unroll
    for (int e = 0; e < DHEAD; ++e){ float ex = __expf(logit[e]-mx); logit[e] = ex; sum += ex; }
    float inv = 1.f / sum;
    #pragma unroll
    for (int e = 0; e < DHEAD; ++e) sA[h][d][e] = logit[e]*inv;
  }
  __syncthreads();
  // W_eff[k=h*28+e][o] = sum_d attn[h][d][e]*proj_w[h*28+d][o]; K padded to 128
  for (int idx = t; idx < KP*DIMC; idx += 128){
    int k = idx / DIMC, o = idx - k*DIMC;
    float s = 0.f;
    if (k < DIMC){
      int h = k / DHEAD, e = k - h*DHEAD;
      #pragma unroll
      for (int d = 0; d < DHEAD; ++d)
        s = fmaf(sA[h][d][e], proj_w[(h*DHEAD + d)*DIMC + o], s);
    }
    int kc, h2, j; kinv(k, kc, h2, j);
    Weff[(long)b*KP*DIMC + (((kc*DIMC) + o)*2 + h2)*16 + j] = f2bf(s);
  }
}

// ---------------- out = v_masked @ W_eff[b] + proj_b (WMMA) ----------------
__global__ __launch_bounds__(256) void out_wmma_kernel(
    const bf16* __restrict__ vmb, const bf16* __restrict__ Weff,
    const float* __restrict__ proj_b, float* __restrict__ out){
  __shared__ bf16 sXp[128 * KP];        // packed [row][kc][hi][j]
  __shared__ bf16 sWp[4 * DIMC * 32];   // packed [kc][o][hi][j]
  const int tid = threadIdx.x, wave = tid >> 5, lane = tid & 31;
  const int hi = lane >> 4, ln = lane & 15;
  const long rowbase = (long)blockIdx.x * 128;
  const int b = (int)(rowbase >> 16);

  // stage masked-v: b32 global loads (bf16 pair) -> b32 packed LDS stores
  for (int idx = tid; idx < 128*56; idx += 256){
    int r = idx / 56, k = (idx - r*56) * 2;
    unsigned g = *reinterpret_cast<const unsigned*>(vmb + (rowbase + r)*DIMC + k);
    int kc, h2, j; kinv(k, kc, h2, j);
    *reinterpret_cast<unsigned*>(&sXp[(((r*4 + kc)*2 + h2)*16) + j]) = g;
  }
  for (int idx = tid; idx < 128*4; idx += 256){   // zero-pad k>=112
    int r = idx >> 2, h2 = (idx >> 1) & 1, qd = idx & 1;
    *reinterpret_cast<unsigned long long*>(
        &sXp[(((r*4 + 3)*2 + h2)*16) + 8 + qd*4]) = 0ull;
  }
  // W_eff already packed: straight b64 copy
  {
    const unsigned long long* src =
        reinterpret_cast<const unsigned long long*>(Weff + (size_t)b*KP*DIMC);
    unsigned long long* dst = reinterpret_cast<unsigned long long*>(&sWp[0]);
    for (int idx = tid; idx < (KP*DIMC)/4; idx += 256) dst[idx] = src[idx];
  }
  __syncthreads();

  v8f acc[7];
  #pragma unroll
  for (int i = 0; i < 7; ++i) acc[i] = (v8f)(0.f);
  const int arow = wave*16 + ln;

  #pragma unroll
  for (int kc = 0; kc < 4; ++kc){
    const v16bf Af = *reinterpret_cast<const v16bf*>(
        &sXp[((arow*4 + kc)*2 + hi)*16]);
    #pragma unroll
    for (int nt = 0; nt < 7; ++nt){
      const v16bf Bf = *reinterpret_cast<const v16bf*>(
          &sWp[(((kc*DIMC) + nt*16 + ln)*2 + hi)*16]);
      acc[nt] = __builtin_amdgcn_wmma_f32_16x16x32_bf16(
          false, Af, false, Bf, (short)0, acc[nt], false, false);
    }
  }
  #pragma unroll
  for (int nt = 0; nt < 7; ++nt){
    int ch = nt*16 + ln;
    float pb = proj_b[ch];
    #pragma unroll
    for (int i = 0; i < 8; ++i){
      long row = rowbase + wave*16 + hi*8 + i;
      out[row*DIMC + ch] = acc[nt][i] + pb;
    }
  }
}

// ---------------- positional branch: dw3x3 -> GELU -> dw3x3 ----------------
__global__ __launch_bounds__(256) void pe1_kernel(
    const bf16* __restrict__ vib, const float* __restrict__ w1, bf16* __restrict__ tb){
  int total = NROWS * DIMC;
  for (int idx = blockIdx.x*blockDim.x + threadIdx.x; idx < total;
       idx += gridDim.x*blockDim.x){
    int ch = idx % DIMC; int row = idx / DIMC;
    int b = row >> 16; int pix = row & (NPIX-1);
    int r = pix >> 8, c = pix & 255;
    float s = 0.f;
    #pragma unroll
    for (int dy = 0; dy < 3; ++dy){
      int rr = r + dy - 1;
      if (rr < 0 || rr >= HH) continue;
      #pragma unroll
      for (int dx = 0; dx < 3; ++dx){
        int c2 = c + dx - 1;
        if (c2 < 0 || c2 >= WW) continue;
        s = fmaf(bf2f(vib[((long)(b<<16 | (rr<<8) | c2))*DIMC + ch]),
                 w1[ch*9 + dy*3 + dx], s);
      }
    }
    float g = 0.5f * s * (1.f + erff(s * 0.70710678f));   // exact GELU
    tb[idx] = f2bf(g);
  }
}

__global__ __launch_bounds__(256) void pe2_kernel(
    const bf16* __restrict__ tb, const float* __restrict__ w2, float* __restrict__ out){
  int total = NROWS * DIMC;
  for (int idx = blockIdx.x*blockDim.x + threadIdx.x; idx < total;
       idx += gridDim.x*blockDim.x){
    int ch = idx % DIMC; int row = idx / DIMC;
    int b = row >> 16; int pix = row & (NPIX-1);
    int r = pix >> 8, c = pix & 255;
    float s = 0.f;
    #pragma unroll
    for (int dy = 0; dy < 3; ++dy){
      int rr = r + dy - 1;
      if (rr < 0 || rr >= HH) continue;
      #pragma unroll
      for (int dx = 0; dx < 3; ++dx){
        int c2 = c + dx - 1;
        if (c2 < 0 || c2 >= WW) continue;
        s = fmaf(bf2f(tb[((long)(b<<16 | (rr<<8) | c2))*DIMC + ch]),
                 w2[ch*9 + dy*3 + dx], s);
      }
    }
    out[idx] += s;
  }
}

// ---------------- host side ----------------
extern "C" void kernel_launch(void* const* d_in, const int* in_sizes, int n_in,
                              void* d_out, int out_size, void* d_ws, size_t ws_size,
                              hipStream_t stream){
  (void)in_sizes; (void)n_in; (void)out_size; (void)ws_size;
  const float* x_in    = (const float*)d_in[0];
  const float* mask    = (const float*)d_in[1];
  const float* Wq      = (const float*)d_in[2];
  const float* Wk      = (const float*)d_in[3];
  const float* Wv      = (const float*)d_in[4];
  const float* rescale = (const float*)d_in[5];
  const float* proj_w  = (const float*)d_in[6];
  const float* proj_b  = (const float*)d_in[7];
  const float* pe_w1   = (const float*)d_in[8];
  const float* pe_w2   = (const float*)d_in[9];
  const float* mm_c1_w = (const float*)d_in[10];
  const float* mm_c1_b = (const float*)d_in[11];
  const float* mm_c2_w = (const float*)d_in[12];
  const float* mm_c2_b = (const float*)d_in[13];
  const float* mm_dw_w = (const float*)d_in[14];
  const float* mm_dw_b = (const float*)d_in[15];
  float* out = (float*)d_out;

  char* w = (char*)d_ws;
  auto alloc = [&](size_t bytes) -> char* {
    char* p = w; w += (bytes + 255) & ~(size_t)255; return p;
  };
  float* m1    = (float*)alloc((size_t)HH*WP2*DIMC*sizeof(float));
  float* m2    = (float*)alloc((size_t)HH*WP2*DIMC*sizeof(float));
  float* mattn = (float*)alloc((size_t)NPIX*DIMC*sizeof(float));
  bf16*  qb    = (bf16*) alloc((size_t)NROWS*DIMC*sizeof(bf16));
  bf16*  kb    = (bf16*) alloc((size_t)NROWS*DIMC*sizeof(bf16));
  bf16*  vmb   = (bf16*) alloc((size_t)NROWS*DIMC*sizeof(bf16));
  bf16*  vib   = (bf16*) alloc((size_t)NROWS*DIMC*sizeof(bf16));
  bf16*  tb    = (bf16*) alloc((size_t)NROWS*DIMC*sizeof(bf16));
  size_t statN = (size_t)BATCH*DIMC*DIMC + 2*(size_t)BATCH*DIMC;
  float* stats = (float*)alloc(statN*sizeof(float));
  float* G     = stats;
  float* normq = stats + (size_t)BATCH*DIMC*DIMC;
  float* normk = normq + (size_t)BATCH*DIMC;
  bf16*  Weff  = (bf16*) alloc((size_t)BATCH*KP*DIMC*sizeof(bf16));

  // mask branch
  conv1x1_kernel <<<4096, 256, 0, stream>>>(mask, mm_c1_w, mm_c1_b, m1, HH*WP2);
  conv1x1_kernel <<<4096, 256, 0, stream>>>(m1,   mm_c2_w, mm_c2_b, m2, HH*WP2);
  maskattn_kernel<<<4096, 256, 0, stream>>>(m1, m2, mm_dw_w, mm_dw_b, mattn);

  hipMemsetAsync(stats, 0, statN*sizeof(float), stream);

  // fused QKV + mask-on-V + column sumsq for norms
  qkv_wmma_kernel<<<NROWS/128, 256, 0, stream>>>(x_in, Wq, Wk, Wv, mattn,
                                                 qb, kb, vmb, vib, normq, normk);
  // Gram K^T Q  (TDM double-buffered staging when available)
  gram_wmma_kernel<<<BATCH*GRAM_SLICES, 224, 0, stream>>>(qb, kb, G);
  // softmax + fold attn into effective projection weight (packed layout)
  attn_weff_kernel<<<BATCH, 128, 0, stream>>>(G, normq, normk, rescale, proj_w, Weff);
  // output GEMM
  out_wmma_kernel<<<NROWS/128, 256, 0, stream>>>(vmb, Weff, proj_b, out);
  // positional branch
  pe1_kernel<<<4096, 256, 0, stream>>>(vib, pe_w1, tb);
  pe2_kernel<<<4096, 256, 0, stream>>>(tb, pe_w2, out);
}
